// HashDecoder_33887291965609
// MI455X (gfx1250) — compile-verified, matrix operand order checked by reading, not run on hardware
//
#include <hip/hip_runtime.h>

typedef __attribute__((ext_vector_type(16))) _Float16 v16h;
typedef __attribute__((ext_vector_type(8)))  _Float16 v8h;
typedef __attribute__((ext_vector_type(2)))  _Float16 v2h;
typedef __attribute__((ext_vector_type(8)))  float    v8f;

#define HD_LEVELS 16
#define HD_TSIZE  (1u << 19)
#define HD_TMASK  (HD_TSIZE - 1u)
#define HD_ROWP   40   // padded LDS row length in f16 (80 bytes, keeps 16B alignment)

#if defined(__has_builtin)
#  if __has_builtin(__builtin_amdgcn_wmma_f16_16x16x32_f16)
#    define HD_F16_OUT 1
#  endif
#endif

__constant__ float hd_scale[HD_LEVELS] = {
  16.f, 21.f, 27.f, 36.f, 48.f, 64.f, 84.f, 111.f,
  147.f, 194.f, 256.f, 337.f, 445.f, 588.f, 776.f, 1024.f
};

// Build a 16-f16 WMMA operand fragment from a 32-f16 LDS row using the
// CDNA5 16-bit A/B layout: lanes 0-15 take K 0-7 and 16-23, lanes 16-31
// take K 8-15 and 24-31 (hi = laneId>>4 selects the pattern).
__device__ __forceinline__ v16h hd_frag(const _Float16* row, int hi) {
  v8h lo = *(const v8h*)(row + hi * 8);
  v8h hc = *(const v8h*)(row + 16 + hi * 8);
  v16h r;
#pragma unroll
  for (int i = 0; i < 8; ++i) { r[i] = lo[i]; r[i + 8] = hc[i]; }
  return r;
}

#ifdef HD_F16_OUT
// relu on a packed-f16 D fragment: llvm.maxnum.v8f16 -> v_pk_max_num_f16 pairs
__device__ __forceinline__ v8h hd_relu8(v8h d) {
  const v8h z = {};
  return __builtin_elementwise_max(d, z);
}
#endif

// One hidden layer for a 16-point tile: D[16x32] = relu(A[16x32] * W[32x32]),
// with D scattered back into the wave-private activation rows in LDS.
__device__ __forceinline__ void hd_hidden_layer(_Float16* fb, const _Float16* arow,
                                                const v16h& b0, const v16h& b1,
                                                int tp, int hi, int nn) {
  const v16h a = hd_frag(arow, hi);
#ifdef HD_F16_OUT
  const v8h cz = {};
  v8h d0 = __builtin_amdgcn_wmma_f16_16x16x32_f16(
      false, a, false, b0, (short)0, cz, false, false);
  v8h d1 = __builtin_amdgcn_wmma_f16_16x16x32_f16(
      false, a, false, b1, (short)0, cz, false, false);
  d0 = hd_relu8(d0);
  d1 = hd_relu8(d1);
#pragma unroll
  for (int e = 0; e < 8; ++e) {          // f16 C/D: element e -> row M = e + 8*hi
    const int pt = tp + e + hi * 8;
    fb[pt * HD_ROWP + nn]      = d0[e];
    fb[pt * HD_ROWP + 16 + nn] = d1[e];
  }
#else
  const v8f cz = {};
  v8f d0 = __builtin_amdgcn_wmma_f32_16x16x32_f16(
      false, a, false, b0, (short)0, cz, false, false);
  v8f d1 = __builtin_amdgcn_wmma_f32_16x16x32_f16(
      false, a, false, b1, (short)0, cz, false, false);
#pragma unroll
  for (int e = 0; e < 8; ++e) {          // f32 C/D: VGPR e -> row M = e + 8*hi
    const int pt = tp + e + hi * 8;
    fb[pt * HD_ROWP + nn]      = (_Float16)fmaxf(d0[e], 0.f);
    fb[pt * HD_ROWP + 16 + nn] = (_Float16)fmaxf(d1[e], 0.f);
  }
#endif
}

__global__ __launch_bounds__(256)
void HashDecoder_33887291965609_kernel(const float* __restrict__ pts,
                                       const float* __restrict__ table,
                                       const float* __restrict__ w1,
                                       const float* __restrict__ w2,
                                       const float* __restrict__ w3,
                                       float* __restrict__ out, int N)
{
  __shared__ _Float16 sW1[32 * 32];         // [out][in] f16, column(out)-major
  __shared__ _Float16 sW2[32 * 32];
  __shared__ _Float16 sW3[16 * 32];         // w3 zero-padded 4 -> 16 outputs
  __shared__ _Float16 sF[8][32 * HD_ROWP];  // per-wave [32 points][32 feats + pad]

  const int tid = threadIdx.x;

  // ---- stage weights -> LDS f16, [out][in] ----
  for (int e = tid; e < 1024; e += 256) {
    const int ic = e & 31, oc = e >> 5;
    sW1[oc * 32 + ic] = (_Float16)w1[ic * 32 + oc];
    sW2[oc * 32 + ic] = (_Float16)w2[ic * 32 + oc];
  }
  for (int e = tid; e < 512; e += 256) {
    const int ic = e & 31, oc = e >> 5;
    sW3[oc * 32 + ic] = (oc < 4) ? (_Float16)w3[ic * 4 + oc] : (_Float16)0.0f;
  }
  __syncthreads();

  const int wave  = tid >> 5;
  const int lane  = tid & 31;
  const int wbase = blockIdx.x * 256 + wave * 32;   // 32 points per wave

  // ---- hash-grid encode: one point per lane, 128 float2 gathers ----
  {
    int pi = wbase + lane;
    if (pi >= N) pi = N - 1;
    // points are read-once streaming data: keep them out of L2 (table owns L2)
    const float px = __builtin_nontemporal_load(&pts[pi * 3 + 0]);
    const float py = __builtin_nontemporal_load(&pts[pi * 3 + 1]);
    const float pz = __builtin_nontemporal_load(&pts[pi * 3 + 2]);
    const float2* __restrict__ tab = (const float2*)table;
    _Float16* frow = &sF[wave][lane * HD_ROWP];

#pragma unroll 8
    for (int l = 0; l < HD_LEVELS; ++l) {
      const float s  = hd_scale[l];
      const float sx = px * s, sy = py * s, sz = pz * s;
      const float fx = floorf(sx), fy = floorf(sy), fz = floorf(sz);
      const float ox = sx - fx, oy = sy - fy, oz = sz - fz;
      const unsigned x0 = (unsigned)fx, y0 = (unsigned)fy, z0 = (unsigned)fz;
      const unsigned x1 = (unsigned)ceilf(sx);
      const unsigned y1 = (unsigned)ceilf(sy);
      const unsigned z1 = (unsigned)ceilf(sz);
      const unsigned lvl = (unsigned)l * HD_TSIZE;
      float f0 = 0.f, f1 = 0.f;
#pragma unroll
      for (int c = 0; c < 8; ++c) {
        // nerfstudio corner order encoded as ceil-bitmasks: x=0x33 y=0x99 z=0x0F
        const bool bx = (0x33u >> c) & 1u;
        const bool by = (0x99u >> c) & 1u;
        const bool bz = (0x0Fu >> c) & 1u;
        const unsigned h = (bx ? x1 : x0)
                         ^ ((by ? y1 : y0) * 2654435761u)
                         ^ ((bz ? z1 : z0) * 805459861u);
        const float2 t = tab[(h & HD_TMASK) + lvl];   // RT hint: live in L2
        const float w = (bx ? ox : 1.f - ox) * (by ? oy : 1.f - oy)
                      * (bz ? oz : 1.f - oz);
        f0 = fmaf(t.x, w, f0);
        f1 = fmaf(t.y, w, f1);
      }
      v2h pk; pk.x = (_Float16)f0; pk.y = (_Float16)f1;
      *(v2h*)&frow[2 * l] = pk;
    }
  }
  // sF writes/reads below are wave-private; LDS ops from one wave are in-order.

  // ---- fused MLP 32->32->32->4 via WMMA ----
  const int hi = lane >> 4;
  const int nn = lane & 15;
  _Float16* fb = sF[wave];

  // Hoist all B (weight) fragments into registers once: stores to sF can't
  // alias sW*, but the compiler can't prove it, so do the CSE by hand.
  const v16h b1a = hd_frag(sW1 + nn * 32, hi);
  const v16h b1b = hd_frag(sW1 + (16 + nn) * 32, hi);
  const v16h b2a = hd_frag(sW2 + nn * 32, hi);
  const v16h b2b = hd_frag(sW2 + (16 + nn) * 32, hi);
  const v16h b3  = hd_frag(sW3 + nn * 32, hi);

#pragma unroll
  for (int tile = 0; tile < 2; ++tile) {
    const int tp = tile * 16;
    const _Float16* arow = fb + (tp + nn) * HD_ROWP;

    hd_hidden_layer(fb, arow, b1a, b1b, tp, hi, nn);   // layer 1 + relu
    hd_hidden_layer(fb, arow, b2a, b2b, tp, hi, nn);   // layer 2 + relu

    // layer 3: f32 accumulate, only cols 0-3 valid (w3 zero-padded); no relu
    const v16h a = hd_frag(arow, hi);
    const v8f cz = {};
    v8f d3 = __builtin_amdgcn_wmma_f32_16x16x32_f16(
        false, a, false, b3, (short)0, cz, false, false);
    if (nn < 4) {                        // EXEC divergence only after all WMMAs
#pragma unroll
      for (int r = 0; r < 8; ++r) {
        const int pt = wbase + tp + r + hi * 8;
        if (pt < N)                      // write-once output: NT store, skip L2
          __builtin_nontemporal_store(d3[r], &out[pt * 4 + nn]);
      }
    }
  }
}

extern "C" void kernel_launch(void* const* d_in, const int* in_sizes, int n_in,
                              void* d_out, int out_size, void* d_ws, size_t ws_size,
                              hipStream_t stream) {
  (void)n_in; (void)out_size; (void)d_ws; (void)ws_size;
  const float* p  = (const float*)d_in[0];
  const float* ht = (const float*)d_in[1];
  const float* w1 = (const float*)d_in[2];
  const float* w2 = (const float*)d_in[3];
  const float* w3 = (const float*)d_in[4];
  float* out = (float*)d_out;
  const int N = in_sizes[0] / 3;
  const int blocks = (N + 255) / 256;
  HashDecoder_33887291965609_kernel<<<blocks, 256, 0, stream>>>(p, ht, w1, w2, w3, out, N);
}
